// CausalSelfAttention_45045617001108
// MI455X (gfx1250) — matrix operand order repaired and checked by reading, not compiled, not run on hardware
//
#include <hip/hip_runtime.h>
#include <hip/hip_bf16.h>

// ---------------------------------------------------------------------------
// CausalSelfAttention for gfx1250 (MI455X): bf16 WMMA + async-to-LDS pipeline
//   B=4, T=2048, C=1024, H=16, D=64.  M = B*T = 8192.
// ---------------------------------------------------------------------------

typedef __bf16 bf16;
typedef __attribute__((ext_vector_type(8)))  bf16  v8bf;
typedef __attribute__((ext_vector_type(16))) bf16  v16bf;
typedef __attribute__((ext_vector_type(8)))  float v8f;

#define T_DIM 2048
#define C_DIM 1024
#define H_DIM 16
#define D_DIM 64
#define N_QKV 3072
#define LDS_STRIDE 40   // bf16 elements per LDS row (80 bytes, 16B aligned)

// 1/sqrt(64) * log2(e): softmax evaluated in base-2 domain (v_exp_f32 native).
#define SCALE_LOG2E 0.1803369940458786f

__device__ __forceinline__ v16bf join16(v8bf lo, v8bf hi) {
    return __builtin_shufflevector(lo, hi, 0,1,2,3,4,5,6,7,8,9,10,11,12,13,14,15);
}

__device__ __forceinline__ v8f wmma_bf16(v16bf a, v16bf b, v8f c) {
    return __builtin_amdgcn_wmma_f32_16x16x32_bf16(false, a, false, b, (short)0, c, false, false);
}

// Async global->LDS copy of 16 bytes per lane (ASYNCcnt-tracked).
__device__ __forceinline__ void async_b128(const bf16* lds_dst, const bf16* gsrc) {
    uint32_t l = (uint32_t)(uintptr_t)lds_dst;              // low 32 bits = LDS offset
    uint64_t g = (uint64_t)(uintptr_t)gsrc;
    asm volatile("global_load_async_to_lds_b128 %0, %1, off" :: "v"(l), "v"(g) : "memory");
}
__device__ __forceinline__ void wait_async0() {
    asm volatile("s_wait_asynccnt 0" ::: "memory");
}

// A fragment (16x32 bf16, MxK) from LDS row-major tile.
// lane<16 -> row=lane, K in {0..7,16..23}; lane>=16 -> row=lane-16, K in {8..15,24..31}.
__device__ __forceinline__ v16bf lds_afrag(const bf16* tile, int row0) {
    int lane = threadIdx.x & 31;
    int row  = row0 + (lane & 15);
    int ko   = (lane < 16) ? 0 : 8;
    const bf16* p = tile + row * LDS_STRIDE + ko;
    return join16(*(const v8bf*)p, *(const v8bf*)(p + 16));
}

// B fragment (32x16 bf16, KxN) from LDS tile stored n-major ([N][K]):
// lane<16 -> n=lane, K in {0..7,16..23}; lane>=16 -> n=lane-16, K in {8..15,24..31}.
__device__ __forceinline__ v16bf lds_bfrag(const bf16* tileT, int n0) {
    int lane = threadIdx.x & 31;
    int n    = n0 + (lane & 15);
    int kr   = (lane < 16) ? 0 : 8;
    const bf16* p = tileT + n * LDS_STRIDE + kr;
    return join16(*(const v8bf*)p, *(const v8bf*)(p + 16));
}

__device__ __forceinline__ float hmax16(float v) {
    v = fmaxf(v, __shfl_xor(v, 1, 32));
    v = fmaxf(v, __shfl_xor(v, 2, 32));
    v = fmaxf(v, __shfl_xor(v, 4, 32));
    v = fmaxf(v, __shfl_xor(v, 8, 32));
    return v;
}
__device__ __forceinline__ float hsum16(float v) {
    v += __shfl_xor(v, 1, 32);
    v += __shfl_xor(v, 2, 32);
    v += __shfl_xor(v, 4, 32);
    v += __shfl_xor(v, 8, 32);
    return v;
}

// ---------------------------------------------------------------------------
// Pre-pass A: fp32 -> bf16 elementwise (x).
// ---------------------------------------------------------------------------
__global__ __launch_bounds__(256)
void f32_to_bf16_kernel(const float* __restrict__ src, bf16* __restrict__ dst)
{
    size_t i = ((size_t)blockIdx.x * 256 + threadIdx.x) * 8;
    float4 a = *(const float4*)(src + i);
    float4 b = *(const float4*)(src + i + 4);
    v8bf o;
    o[0] = (bf16)a.x; o[1] = (bf16)a.y; o[2] = (bf16)a.z; o[3] = (bf16)a.w;
    o[4] = (bf16)b.x; o[5] = (bf16)b.y; o[6] = (bf16)b.z; o[7] = (bf16)b.w;
    *(v8bf*)(dst + i) = o;
}

// ---------------------------------------------------------------------------
// Pre-pass B: W[K][N] fp32 -> WT[N][K] bf16 (tiled 32x32 LDS transpose).
// ---------------------------------------------------------------------------
__global__ __launch_bounds__(256)
void transpose_conv_kernel(const float* __restrict__ W, bf16* __restrict__ WT, int N)
{
    __shared__ float t[32][33];
    const int n0 = blockIdx.x * 32;
    const int k0 = blockIdx.y * 32;
    const int tn = threadIdx.x & 31;
    const int tk = threadIdx.x >> 5;   // 0..7
    #pragma unroll
    for (int p = 0; p < 4; ++p)
        t[tk + p * 8][tn] = W[(size_t)(k0 + tk + p * 8) * N + n0 + tn];
    __syncthreads();
    #pragma unroll
    for (int p = 0; p < 4; ++p)
        WT[(size_t)(n0 + tk + p * 8) * C_DIM + k0 + tn] = (bf16)t[tn][tk + p * 8];
}

// ---------------------------------------------------------------------------
// Double-buffered async GEMM: C = A(bf16,[M][1024]) @ BT(bf16,[N][1024])^T.
// Block: 256 threads = 8 waves (4x2), tile 128x128, K-step 32.
// MODE 0: qkv epilogue (bias + scatter to K/Q/V head layouts, V transposed).
// MODE 1: proj epilogue (bias + fp32 out).
// ---------------------------------------------------------------------------
template<int MODE>
__global__ __launch_bounds__(256)
void gemm_async_kernel(const bf16* __restrict__ A, const bf16* __restrict__ BT,
                       const float* __restrict__ bias,
                       bf16* __restrict__ Kh, bf16* __restrict__ Qh,
                       bf16* __restrict__ Vt, float* __restrict__ OF)
{
    __shared__ bf16 As[2][128 * LDS_STRIDE];
    __shared__ bf16 Bs[2][128 * LDS_STRIDE];

    const int tid  = threadIdx.x;
    const int wave = tid >> 5;
    const int lane = tid & 31;
    const int wm   = wave >> 1;            // 0..3 (rows of 32)
    const int wn   = wave & 1;             // 0..1 (cols of 64)
    const int mtile = blockIdx.y * 128;
    const int ntile = blockIdx.x * 128;

    // Each thread owns two 16B chunks of each 128x32 tile (512 chunks total).
    auto issue_tile = [&](int bufsel, int k0) {
        #pragma unroll
        for (int i = 0; i < 2; ++i) {
            int idx = tid * 2 + i;
            int row = idx >> 2;
            int q8  = (idx & 3) * 8;       // bf16 elements
            async_b128(&As[bufsel][row * LDS_STRIDE + q8],
                       A + (size_t)(mtile + row) * C_DIM + k0 + q8);
            async_b128(&Bs[bufsel][row * LDS_STRIDE + q8],
                       BT + (size_t)(ntile + row) * C_DIM + k0 + q8);
        }
    };

    v8f acc[2][4] = {};
    issue_tile(0, 0);
    int buf = 0;

    for (int ki = 0; ki < C_DIM / 32; ++ki) {
        wait_async0();
        __syncthreads();                    // tile ki resident; buf^1 fully drained
        if (ki + 1 < C_DIM / 32) issue_tile(buf ^ 1, (ki + 1) * 32);

        v16bf af[2], bfg[4];
        #pragma unroll
        for (int fm = 0; fm < 2; ++fm) af[fm] = lds_afrag(As[buf], wm * 32 + fm * 16);
        #pragma unroll
        for (int fn = 0; fn < 4; ++fn) bfg[fn] = lds_bfrag(Bs[buf], wn * 64 + fn * 16);
        #pragma unroll
        for (int fm = 0; fm < 2; ++fm)
            #pragma unroll
            for (int fn = 0; fn < 4; ++fn)
                acc[fm][fn] = wmma_bf16(af[fm], bfg[fn], acc[fm][fn]);
        buf ^= 1;
    }

    const int half8 = (lane >> 4) * 8;
    const int colL  = lane & 15;

    if constexpr (MODE == 0) {
        // Section (k/q/v) is uniform per block: 128-wide tiles never straddle 1024.
        const int sect = ntile >> 10;                   // 0=k,1=q,2=v
        const int ci0  = ntile & 1023;
        bf16* KQ = (sect == 0) ? Kh : Qh;
        #pragma unroll
        for (int fn = 0; fn < 4; ++fn) {
            int ci = ci0 + wn * 64 + fn * 16 + colL;
            float bv = bias[ntile + wn * 64 + fn * 16 + colL];
            int h = ci >> 6, d = ci & 63;
            #pragma unroll
            for (int fm = 0; fm < 2; ++fm) {
                int mbase = mtile + wm * 32 + fm * 16 + half8;
                #pragma unroll
                for (int r = 0; r < 8; ++r) {
                    int m = mbase + r;
                    int b = m >> 11;
                    int t = m & (T_DIM - 1);
                    bf16 o = (bf16)(acc[fm][fn][r] + bv);
                    size_t bh = (size_t)(b * H_DIM + h);
                    if (sect < 2) KQ[(bh * T_DIM + t) * D_DIM + d] = o;
                    else          Vt[(bh * D_DIM + d) * T_DIM + t] = o;
                }
            }
        }
    } else {
        #pragma unroll
        for (int fn = 0; fn < 4; ++fn) {
            int n = ntile + wn * 64 + fn * 16 + colL;
            float bv = bias[n];
            #pragma unroll
            for (int fm = 0; fm < 2; ++fm) {
                int mbase = mtile + wm * 32 + fm * 16 + half8;
                #pragma unroll
                for (int r = 0; r < 8; ++r)
                    OF[(size_t)(mbase + r) * C_DIM + n] = acc[fm][fn][r] + bv;
            }
        }
    }
}

// ---------------------------------------------------------------------------
// Flash attention.  grid = (T/128, B*H), 8 waves/block, one wave owns a
// 16-row Q tile and streams KV in chunks of 32.  KV loop split into an
// interior phase (no masking/clamps) and a boundary phase (diagonal tiles).
// Softmax runs in the base-2 domain; row-sums are per-lane partials reduced
// once at the end.
// ---------------------------------------------------------------------------
__global__ __launch_bounds__(256)
void attn_kernel(const bf16* __restrict__ Qh, const bf16* __restrict__ Kh,
                 const bf16* __restrict__ Vt, const int* __restrict__ pmask,
                 bf16* __restrict__ Y)
{
    __shared__ bf16 Plds[8][16 * LDS_STRIDE];

    const int tid  = threadIdx.x;
    const int w    = tid >> 5;
    const int lane = tid & 31;
    const int bh   = blockIdx.y;
    const int b    = bh >> 4;
    const int h    = bh & 15;
    const int q0   = blockIdx.x * 128 + w * 16;

    const bf16* qp = Qh + (size_t)bh * T_DIM * D_DIM;
    const bf16* kp = Kh + (size_t)bh * T_DIM * D_DIM;
    const bf16* vp = Vt + (size_t)bh * D_DIM * T_DIM;

    const int half8 = (lane >> 4) * 8;
    const int colL  = lane & 15;

    v16bf qf[2];
    {
        int row = q0 + (lane & 15);
        int ko  = (lane < 16) ? 0 : 8;
        #pragma unroll
        for (int c = 0; c < 2; ++c) {
            const bf16* p = qp + (size_t)row * D_DIM + c * 32 + ko;
            qf[c] = join16(*(const v8bf*)p, *(const v8bf*)(p + 16));
        }
    }

    float mrun[8], lrun[8];       // lrun: per-lane partial row sums
    v8f acc[4] = {};
    #pragma unroll
    for (int r = 0; r < 8; ++r) { mrun[r] = -1e30f; lrun[r] = 0.0f; }

    auto do_chunk = [&](int kc, bool boundary) {
        // ---- S = Q @ K^T (16 x 32), two 16-col fragments ----
        v8f S[2] = {};
        #pragma unroll
        for (int nb = 0; nb < 2; ++nb) {
            int kv = kc + nb * 16 + colL;
            if (boundary && kv > T_DIM - 1) kv = T_DIM - 1;
            #pragma unroll
            for (int c = 0; c < 2; ++c) {
                int d0 = c * 32 + ((lane < 16) ? 0 : 8);
                const bf16* p = kp + (size_t)kv * D_DIM + d0;
                v16bf kf = join16(*(const v8bf*)p, *(const v8bf*)(p + 16));
                S[nb] = wmma_bf16(qf[c], kf, S[nb]);
            }
        }

        // ---- scale (base-2) + masks ----
        int pmv[2];
        #pragma unroll
        for (int nb = 0; nb < 2; ++nb) {
            int col = kc + nb * 16 + colL;
            pmv[nb] = (!boundary || col < T_DIM) ? pmask[b * T_DIM + (boundary ? (col < T_DIM ? col : T_DIM - 1) : col)] : 0;
            if (boundary && col >= T_DIM) pmv[nb] = 0;
        }
        #pragma unroll
        for (int nb = 0; nb < 2; ++nb)
            #pragma unroll
            for (int r = 0; r < 8; ++r) {
                float s = S[nb][r] * SCALE_LOG2E;
                if (boundary) {
                    int col = kc + nb * 16 + colL;
                    int row = q0 + half8 + r;
                    if (col > row || pmv[nb] == 0) s = -1e30f;
                } else {
                    if (pmv[nb] == 0) s = -1e30f;
                }
                S[nb][r] = s;
            }

        // ---- online softmax (base-2); lrun kept as per-lane partial ----
        #pragma unroll
        for (int r = 0; r < 8; ++r) {
            float mr = hmax16(fmaxf(S[0][r], S[1][r]));
            float mn = fmaxf(mrun[r], mr);
            float al = exp2f(mrun[r] - mn);
            float p0 = exp2f(S[0][r] - mn);
            float p1 = exp2f(S[1][r] - mn);
            lrun[r] = lrun[r] * al + (p0 + p1);
            mrun[r] = mn;
            S[0][r] = p0; S[1][r] = p1;
            #pragma unroll
            for (int c = 0; c < 4; ++c) acc[c][r] *= al;
        }

        // ---- P: C-layout f32 -> A-layout bf16 via per-wave LDS bounce ----
        bf16* pl = Plds[w];
        #pragma unroll
        for (int r = 0; r < 8; ++r) {
            pl[(half8 + r) * LDS_STRIDE + colL]      = (bf16)S[0][r];
            pl[(half8 + r) * LDS_STRIDE + 16 + colL] = (bf16)S[1][r];
        }
        asm volatile("s_wait_dscnt 0" ::: "memory");
        v16bf pf;
        {
            int row = lane & 15;
            int ko  = (lane < 16) ? 0 : 8;
            const bf16* p = pl + row * LDS_STRIDE + ko;
            pf = join16(*(const v8bf*)p, *(const v8bf*)(p + 16));
        }

        // ---- acc += P @ V (V transposed: contiguous B-frags) ----
        #pragma unroll
        for (int c = 0; c < 4; ++c) {
            int d  = c * 16 + colL;
            int kb = kc + ((lane < 16) ? 0 : 8);
            if (boundary && kb >= T_DIM) kb = T_DIM - 32;
            const bf16* p = vp + (size_t)d * T_DIM + kb;
            v16bf vf = join16(*(const v8bf*)p, *(const v8bf*)(p + 16));
            acc[c] = wmma_bf16(pf, vf, acc[c]);
        }
        asm volatile("s_wait_dscnt 0" ::: "memory");
    };

    int kc = 0;
    for (; kc + 32 <= q0; kc += 32) do_chunk(kc, false);   // strictly-lower tiles
    for (; kc < q0 + 16; kc += 32) do_chunk(kc, true);     // diagonal tiles

    // ---- final row-sum reduction + normalize + write y (bf16) ----
    #pragma unroll
    for (int r = 0; r < 8; ++r) {
        float inv = 1.0f / hsum16(lrun[r]);
        int row = q0 + half8 + r;
        size_t base = ((size_t)b * T_DIM + row) * C_DIM + h * D_DIM;
        #pragma unroll
        for (int c = 0; c < 4; ++c)
            Y[base + c * 16 + colL] = (bf16)(acc[c][r] * inv);
    }
}

// ---------------------------------------------------------------------------
extern "C" void kernel_launch(void* const* d_in, const int* in_sizes, int n_in,
                              void* d_out, int out_size, void* d_ws, size_t ws_size,
                              hipStream_t stream)
{
    const float* x      = (const float*)d_in[0];
    const float* W_kqv  = (const float*)d_in[1];
    const float* b_kqv  = (const float*)d_in[2];
    const float* W_proj = (const float*)d_in[3];
    const float* b_proj = (const float*)d_in[4];
    const int*   pmask  = (const int*)d_in[5];

    const size_t nqkv = (size_t)4 * H_DIM * T_DIM * D_DIM;   // 8,388,608
    bf16* xbf    = (bf16*)d_ws;                 // [8192][1024]
    bf16* wt_kqv = xbf + nqkv;                  // [3072][1024]
    bf16* wt_prj = wt_kqv + (size_t)N_QKV * C_DIM;   // [1024][1024]
    bf16* q_ws   = wt_prj + (size_t)C_DIM * C_DIM;
    bf16* k_ws   = q_ws + nqkv;
    bf16* v_ws   = k_ws + nqkv;                 // transposed [B,H,D,T]
    bf16* y_ws   = v_ws + nqkv;                 // [B,T,C]

    f32_to_bf16_kernel<<<dim3(nqkv / (256 * 8)), 256, 0, stream>>>(x, xbf);
    transpose_conv_kernel<<<dim3(N_QKV / 32, C_DIM / 32), 256, 0, stream>>>(W_kqv, wt_kqv, N_QKV);
    transpose_conv_kernel<<<dim3(C_DIM / 32, C_DIM / 32), 256, 0, stream>>>(W_proj, wt_prj, C_DIM);

    gemm_async_kernel<0><<<dim3(N_QKV / 128, 8192 / 128), 256, 0, stream>>>(
        xbf, wt_kqv, b_kqv, k_ws, q_ws, v_ws, nullptr);

    attn_kernel<<<dim3(T_DIM / 128, 4 * H_DIM), 256, 0, stream>>>(
        q_ws, k_ws, v_ws, pmask, y_ws);

    gemm_async_kernel<1><<<dim3(C_DIM / 128, 8192 / 128), 256, 0, stream>>>(
        y_ws, wt_prj, b_proj, nullptr, nullptr, nullptr, (float*)d_out);
}
